// PointNetEmbedding_4801773437700
// MI455X (gfx1250) — matrix operand order repaired and checked by reading, not compiled, not run on hardware
//
#include <hip/hip_runtime.h>
#include <hip/hip_bf16.h>

typedef _Float16 half_t;
typedef __attribute__((ext_vector_type(16))) _Float16 v16h;
typedef __attribute__((ext_vector_type(8)))  _Float16 v8h;
typedef __attribute__((ext_vector_type(8)))  float    v8f;

#define KNB 33            // k+1 neighbors
#define RAD2 0.04f        // R^2 = 0.2^2

// ---------------- elementwise converts ----------------
__global__ void k_f32_to_f16(const float* __restrict__ in, half_t* __restrict__ out, long n) {
  long stride = (long)gridDim.x * blockDim.x;
  for (long g = (long)blockIdx.x * blockDim.x + threadIdx.x; g < n; g += stride)
    out[g] = (half_t)in[g];
}
__global__ void k_f16_to_f32(const half_t* __restrict__ in, float* __restrict__ out, long n) {
  long stride = (long)gridDim.x * blockDim.x;
  for (long g = (long)blockIdx.x * blockDim.x + threadIdx.x; g < n; g += stride)
    out[g] = (float)in[g];
}

// --------- weight prep: W[K,N] f32 -> Wt[N,Kp] f16, zero-padded K ----------
__global__ void k_prep_w(const float* __restrict__ W, int K, int N, int Kp,
                         half_t* __restrict__ Wt) {
  long total = (long)N * Kp;
  long stride = (long)gridDim.x * blockDim.x;
  for (long g = (long)blockIdx.x * blockDim.x + threadIdx.x; g < total; g += stride) {
    int n = (int)(g / Kp), k = (int)(g % Kp);
    Wt[g] = (k < K) ? (half_t)W[(size_t)k * N + n] : (half_t)0.f;
  }
}

// ---------------- farthest point sampling (1 block per batch) ----------------
__global__ void k_fps(const float* __restrict__ posAll, int n, int m,
                      int* __restrict__ idxAll, float* __restrict__ distAll) {
  int b = blockIdx.x;
  const float* pos = posAll + (long)b * n * 3;
  float* dist = distAll + (long)b * n;
  int* idx = idxAll + (long)b * m;
  int tid = threadIdx.x, bs = blockDim.x;
  __shared__ float sd[256];
  __shared__ int   si[256];
  for (int j = tid; j < n; j += bs) dist[j] = 1e30f;
  if (tid == 0) idx[0] = 0;
  __syncthreads();
  int last = 0;
  for (int i = 1; i < m; ++i) {
    float px = pos[last*3+0], py = pos[last*3+1], pz = pos[last*3+2];
    float bd = -1.f; int bi = 0;
    for (int j = tid; j < n; j += bs) {
      float dx = pos[j*3+0]-px, dy = pos[j*3+1]-py, dz = pos[j*3+2]-pz;
      float d = dx*dx + dy*dy + dz*dz;
      float nd = fminf(dist[j], d);
      dist[j] = nd;
      if (nd > bd || (nd == bd && j < bi)) { bd = nd; bi = j; }   // argmax, first-on-tie
    }
    sd[tid] = bd; si[tid] = bi;
    __syncthreads();
    for (int off = bs >> 1; off > 0; off >>= 1) {
      if (tid < off) {
        if (sd[tid+off] > sd[tid] || (sd[tid+off] == sd[tid] && si[tid+off] < si[tid])) {
          sd[tid] = sd[tid+off]; si[tid] = si[tid+off];
        }
      }
      __syncthreads();
    }
    last = si[0];
    if (tid == 0) idx[i] = last;
    __syncthreads();
  }
}

// ---------------- gather center positions ----------------
__global__ void k_gather_cpos(const float* __restrict__ pos, const int* __restrict__ idx,
                              int n, int m, int total, float* __restrict__ cpos) {
  int g = blockIdx.x * blockDim.x + threadIdx.x;
  if (g >= total) return;            // g = b*m + j
  int b = g / m;
  int id = idx[g];
  const float* p = pos + ((long)b * n + id) * 3;
  cpos[(long)g*3+0] = p[0]; cpos[(long)g*3+1] = p[1]; cpos[(long)g*3+2] = p[2];
}

// ---------------- kNN (33 smallest d^2) + radius mask (1 block per center) ---
__global__ void k_knn(const float* __restrict__ pos, const float* __restrict__ cpos,
                      int n, int m, int* __restrict__ nidx, float* __restrict__ mask) {
  int center = blockIdx.x;           // b*m + j
  int b = center / m;
  const float* P = pos + (long)b * n * 3;
  float cx = cpos[center*3+0], cy = cpos[center*3+1], cz = cpos[center*3+2];
  int tid = threadIdx.x, bs = blockDim.x;
  __shared__ float sd[128];
  __shared__ int   si[128];
  float ld = -1.f; int li = -1;      // last selected key (d2, idx), lexicographic
  for (int t = 0; t < KNB; ++t) {
    float bd = 1e30f; int bi = n;
    for (int p = tid; p < n; p += bs) {
      float dx = P[p*3+0]-cx, dy = P[p*3+1]-cy, dz = P[p*3+2]-cz;
      float d = dx*dx + dy*dy + dz*dz;
      bool cand = (d > ld) || (d == ld && p > li);
      if (cand && (d < bd || (d == bd && p < bi))) { bd = d; bi = p; }
    }
    sd[tid] = bd; si[tid] = bi;
    __syncthreads();
    for (int off = bs >> 1; off > 0; off >>= 1) {
      if (tid < off) {
        if (sd[tid+off] < sd[tid] || (sd[tid+off] == sd[tid] && si[tid+off] < si[tid])) {
          sd[tid] = sd[tid+off]; si[tid] = si[tid+off];
        }
      }
      __syncthreads();
    }
    ld = sd[0]; li = si[0];
    if (tid == 0) {
      nidx[(long)center*KNB + t] = li;
      mask[(long)center*KNB + t] = (ld <= RAD2) ? 1.f : 0.f;
    }
    __syncthreads();
  }
}

// -------- build neighbor features [rows, Cfp] f16, zero-padded cols ---------
__global__ void k_gather_feat(const half_t* __restrict__ x, int Cx,
                              const float* __restrict__ pos, const float* __restrict__ cpos,
                              const int* __restrict__ nidx, int n, int m, int Cfp,
                              half_t* __restrict__ feat, long total) {
  int Cf = Cx + 3;
  long stride = (long)gridDim.x * blockDim.x;
  for (long g = (long)blockIdx.x * blockDim.x + threadIdx.x; g < total; g += stride) {
    long row = g / Cfp; int c = (int)(g % Cfp);
    long center = row / KNB; int k = (int)(row % KNB);
    int b = (int)(center / m);
    int nid = nidx[center*KNB + k];
    half_t v;
    if (c < Cx) v = x[((long)b*n + nid)*Cx + c];
    else if (c < Cf) {
      int d = c - Cx;
      v = (half_t)(pos[((long)b*n + nid)*3 + d] - cpos[center*3 + d]);
    } else v = (half_t)0.f;
    feat[g] = v;
  }
}

// ---------------- masked max over KNB neighbors ----------------
__global__ void k_maxagg(const half_t* __restrict__ h, const float* __restrict__ mask,
                         int C, long centers, half_t* __restrict__ out) {
  long total = centers * C;
  long stride = (long)gridDim.x * blockDim.x;
  for (long g = (long)blockIdx.x * blockDim.x + threadIdx.x; g < total; g += stride) {
    long center = g / C; int c = (int)(g % C);
    float best = -1e30f;                    // self (d=0) is always unmasked
    for (int k = 0; k < KNB; ++k)
      if (mask[center*KNB + k] > 0.5f)
        best = fmaxf(best, (float)h[(center*KNB + k)*C + c]);
    out[g] = (half_t)best;
  }
}

// -------- concat [h(512) | cpos(3) | zero-pad] -> stride 544 ----------------
__global__ void k_concat_pos(const half_t* __restrict__ h, const float* __restrict__ cpos,
                             int Cin, int Cfp, long rows, half_t* __restrict__ out) {
  long total = rows * Cfp;
  long stride = (long)gridDim.x * blockDim.x;
  for (long g = (long)blockIdx.x * blockDim.x + threadIdx.x; g < total; g += stride) {
    long r = g / Cfp; int c = (int)(g % Cfp);
    half_t v;
    if (c < Cin) v = h[r*Cin + c];
    else if (c < Cin + 3) v = (half_t)cpos[r*3 + (c - Cin)];
    else v = (half_t)0.f;
    out[g] = v;
  }
}

// ---------------- global max pool over m centers ----------------
__global__ void k_maxpool(const half_t* __restrict__ in, int m, int C, int Bn,
                          half_t* __restrict__ out) {
  int g = blockIdx.x * blockDim.x + threadIdx.x;
  int total = Bn * C;
  if (g >= total) return;
  int b = g / C, c = g % C;
  float best = -1e30f;
  for (int j = 0; j < m; ++j)
    best = fmaxf(best, (float)in[((long)b*m + j)*C + c]);
  out[g] = (half_t)best;
}

// ---- WMMA GEMM (fast path): M % 32 == 0, no row guards ---------------------
// C[M,N] = act(A_f16[M,Kp] @ Wt_f16[N,Kp]^T + bias).
// block = 128 threads (4 wave32s). Block tile = 32 rows x 64 cols.
// A tile staged via CDNA5 async global->LDS DMA (ASYNCcnt), B frags straight
// from global (weights are L2-hot). Two WMMAs per K-step share one B frag.
__global__ void k_wmma_gemm(const half_t* __restrict__ A, const half_t* __restrict__ Wt,
                            const float* __restrict__ bias, half_t* __restrict__ C,
                            int M, int N, int Kp, int relu) {
  __shared__ half_t sA[32][32];   // only LDS object -> allocated at LDS offset 0
  int tid  = threadIdx.x;
  int wave = tid >> 5, lane = tid & 31;
  int row0 = blockIdx.x * 32;
  int col0 = blockIdx.y * 64 + wave * 16;
  int ar  = lane & 15, akb = (lane >> 4) * 8;    // A frag: row, K-subblock
  int bc  = col0 + (lane & 15);                  // B frag column
  int bkb = (lane >> 4) * 8;                     // B frag K-subblock
  int ldr = tid >> 2, ldk = (tid & 3) * 8;       // A stage: 1 x b128 per thread
  unsigned ldsa = (unsigned)((ldr * 32 + ldk) * (int)sizeof(half_t)); // byte off in LDS
  const half_t* gp = A + (size_t)(row0 + ldr) * Kp + ldk;
  v8f acc0 = {}, acc1 = {};
  for (int k0 = 0; k0 < Kp; k0 += 32) {
    // CDNA5 async DMA: 16B global -> LDS per thread, tracked by ASYNCcnt
    asm volatile("global_load_async_to_lds_b128 %0, %1, off"
                 :: "v"(ldsa), "v"(gp + k0) : "memory");
    asm volatile("s_wait_asynccnt 0x0" ::: "memory");
    __syncthreads();
    // B fragment: halves [bkb,bkb+8) and [16+bkb,16+bkb+8) per lane, straight from global (L2-hot)
    const half_t* wp = Wt + (size_t)bc * Kp + k0 + bkb;
    v8h bl = *(const v8h*)wp;
    v8h bh = *(const v8h*)(wp + 16);
    v16h bf = __builtin_shufflevector(bl, bh, 0,1,2,3,4,5,6,7,8,9,10,11,12,13,14,15);
    // A fragments: ds_load_b128 pairs
    v8h a0l = *(const v8h*)&sA[ar][akb];
    v8h a0h = *(const v8h*)&sA[ar][16 + akb];
    v16h af0 = __builtin_shufflevector(a0l, a0h, 0,1,2,3,4,5,6,7,8,9,10,11,12,13,14,15);
    v8h a1l = *(const v8h*)&sA[16 + ar][akb];
    v8h a1h = *(const v8h*)&sA[16 + ar][16 + akb];
    v16h af1 = __builtin_shufflevector(a1l, a1h, 0,1,2,3,4,5,6,7,8,9,10,11,12,13,14,15);
    acc0 = __builtin_amdgcn_wmma_f32_16x16x32_f16(false, af0, false, bf, (short)0, acc0,
                                                  false, false);
    acc1 = __builtin_amdgcn_wmma_f32_16x16x32_f16(false, af1, false, bf, (short)0, acc1,
                                                  false, false);
    __syncthreads();   // prior frag reads done before next async overwrite
  }
  int col = col0 + (lane & 15);
  int rb  = row0 + (lane >> 4) * 8;
  float bv = bias[col];
  #pragma unroll
  for (int v = 0; v < 8; ++v) {
    float v0 = acc0[v] + bv, v1 = acc1[v] + bv;
    if (relu) { v0 = fmaxf(v0, 0.f); v1 = fmaxf(v1, 0.f); }
    C[(size_t)(rb + v) * N + col]      = (half_t)v0;
    C[(size_t)(rb + 16 + v) * N + col] = (half_t)v1;
  }
}

// ---- WMMA GEMM (guarded path): arbitrary M (used for the final M=8 GEMMs) --
__global__ void k_wmma_gemm_g(const half_t* __restrict__ A, const half_t* __restrict__ Wt,
                              const float* __restrict__ bias, half_t* __restrict__ C,
                              int M, int N, int Kp, int relu) {
  __shared__ half_t sA[32][32];
  int tid  = threadIdx.x;
  int wave = tid >> 5, lane = tid & 31;
  int row0 = blockIdx.x * 32;
  int col0 = blockIdx.y * 64 + wave * 16;
  int ar  = lane & 15, akb = (lane >> 4) * 8;
  int bc  = col0 + (lane & 15);
  int bkb = (lane >> 4) * 8;
  int ldr = tid >> 2, ldk = (tid & 3) * 8;
  v8f acc0 = {}, acc1 = {};
  for (int k0 = 0; k0 < Kp; k0 += 32) {
    {
      int gr = row0 + ldr;
      v8h av = {};
      if (gr < M) av = *(const v8h*)(A + (size_t)gr * Kp + k0 + ldk);
      *(v8h*)&sA[ldr][ldk] = av;
    }
    __syncthreads();
    const half_t* wp = Wt + (size_t)bc * Kp + k0 + bkb;
    v8h bl = *(const v8h*)wp;
    v8h bh = *(const v8h*)(wp + 16);
    v16h bf = __builtin_shufflevector(bl, bh, 0,1,2,3,4,5,6,7,8,9,10,11,12,13,14,15);
    v8h a0l = *(const v8h*)&sA[ar][akb];
    v8h a0h = *(const v8h*)&sA[ar][16 + akb];
    v16h af0 = __builtin_shufflevector(a0l, a0h, 0,1,2,3,4,5,6,7,8,9,10,11,12,13,14,15);
    v8h a1l = *(const v8h*)&sA[16 + ar][akb];
    v8h a1h = *(const v8h*)&sA[16 + ar][16 + akb];
    v16h af1 = __builtin_shufflevector(a1l, a1h, 0,1,2,3,4,5,6,7,8,9,10,11,12,13,14,15);
    acc0 = __builtin_amdgcn_wmma_f32_16x16x32_f16(false, af0, false, bf, (short)0, acc0,
                                                  false, false);
    acc1 = __builtin_amdgcn_wmma_f32_16x16x32_f16(false, af1, false, bf, (short)0, acc1,
                                                  false, false);
    __syncthreads();
  }
  int col = col0 + (lane & 15);
  int rb  = row0 + (lane >> 4) * 8;
  float bv = bias[col];
  #pragma unroll
  for (int v = 0; v < 8; ++v) {
    int r = rb + v;
    if (r < M) {
      float val = acc0[v] + bv;
      if (relu) val = fmaxf(val, 0.f);
      C[(size_t)r * N + col] = (half_t)val;
    }
  }
  #pragma unroll
  for (int v = 0; v < 8; ++v) {
    int r = rb + 16 + v;
    if (r < M) {
      float val = acc1[v] + bv;
      if (relu) val = fmaxf(val, 0.f);
      C[(size_t)r * N + col] = (half_t)val;
    }
  }
}

// =====================================================================

static inline unsigned gs1(long total) {
  long b = (total + 255) / 256;
  if (b < 1) b = 1;
  if (b > 1048576) b = 1048576;
  return (unsigned)b;
}

extern "C" void kernel_launch(void* const* d_in, const int* in_sizes, int n_in,
                              void* d_out, int out_size, void* d_ws, size_t ws_size,
                              hipStream_t stream) {
  const int Bn = 8, N0 = 8192, m1 = 2048, m2 = 512, m3 = 128;

  const float* x   = (const float*)d_in[0];
  const float* pos = (const float*)d_in[1];
  // d_in[2] = batch (implicit in dense layout, unused)
  const float* Wl1_0 = (const float*)d_in[3];  const float* bl1_0 = (const float*)d_in[4];
  const float* Wl1_1 = (const float*)d_in[5];  const float* bl1_1 = (const float*)d_in[6];
  const float* Wl1_2 = (const float*)d_in[7];  const float* bl1_2 = (const float*)d_in[8];
  const float* Wg1   = (const float*)d_in[9];  const float* bg1   = (const float*)d_in[10];
  const float* Wl2_0 = (const float*)d_in[11]; const float* bl2_0 = (const float*)d_in[12];
  const float* Wl2_1 = (const float*)d_in[13]; const float* bl2_1 = (const float*)d_in[14];
  const float* Wl2_2 = (const float*)d_in[15]; const float* bl2_2 = (const float*)d_in[16];
  const float* Wg2   = (const float*)d_in[17]; const float* bg2   = (const float*)d_in[18];
  const float* Wl3_0 = (const float*)d_in[19]; const float* bl3_0 = (const float*)d_in[20];
  const float* Wl3_1 = (const float*)d_in[21]; const float* bl3_1 = (const float*)d_in[22];
  const float* Wg3   = (const float*)d_in[23]; const float* bg3   = (const float*)d_in[24];
  const float* Wgs_0 = (const float*)d_in[25]; const float* bgs_0 = (const float*)d_in[26];
  const float* Wgs_1 = (const float*)d_in[27]; const float* bgs_1 = (const float*)d_in[28];
  const float* Wf_0  = (const float*)d_in[29]; const float* bf_0  = (const float*)d_in[30];
  const float* Wf_1  = (const float*)d_in[31]; const float* bf_1  = (const float*)d_in[32];

  // ---- workspace bump allocator ----
  char* base = (char*)d_ws; size_t off = 0;
  auto alloc = [&](size_t bytes) -> void* {
    void* p = base + off; off = (off + bytes + 255) & ~(size_t)255; return p;
  };
  half_t* xh0   = (half_t*)alloc((size_t)Bn*N0*3 * sizeof(half_t));
  float*  dist  = (float*) alloc((size_t)Bn*N0 * sizeof(float));
  int*    idx1  = (int*)   alloc((size_t)Bn*m1 * sizeof(int));
  float*  cpos1 = (float*) alloc((size_t)Bn*m1*3 * sizeof(float));
  int*    nidx1 = (int*)   alloc((size_t)Bn*m1*KNB * sizeof(int));
  float*  mask1 = (float*) alloc((size_t)Bn*m1*KNB * sizeof(float));
  int*    idx2  = (int*)   alloc((size_t)Bn*m2 * sizeof(int));
  float*  cpos2 = (float*) alloc((size_t)Bn*m2*3 * sizeof(float));
  int*    nidx2 = (int*)   alloc((size_t)Bn*m2*KNB * sizeof(int));
  float*  mask2 = (float*) alloc((size_t)Bn*m2*KNB * sizeof(float));
  int*    idx3  = (int*)   alloc((size_t)Bn*m3 * sizeof(int));
  float*  cpos3 = (float*) alloc((size_t)Bn*m3*3 * sizeof(float));
  int*    nidx3 = (int*)   alloc((size_t)Bn*m3*KNB * sizeof(int));
  float*  mask3 = (float*) alloc((size_t)Bn*m3*KNB * sizeof(float));
  half_t* x2    = (half_t*)alloc((size_t)Bn*m1*128 * sizeof(half_t));
  half_t* x3    = (half_t*)alloc((size_t)Bn*m2*256 * sizeof(half_t));
  half_t* hg    = (half_t*)alloc((size_t)Bn*m3*512 * sizeof(half_t));
  half_t* pooled= (half_t*)alloc((size_t)Bn*1024 * sizeof(half_t));
  half_t* t1    = (half_t*)alloc((size_t)Bn*512 * sizeof(half_t));
  half_t* t2    = (half_t*)alloc((size_t)Bn*256 * sizeof(half_t));

  // f16 transposed + K-padded weights (~4.2 MB total, L2-resident)
  auto prep = [&](const float* W, int K, int N, int Kp) -> half_t* {
    half_t* Wt = (half_t*)alloc((size_t)N * Kp * sizeof(half_t));
    k_prep_w<<<gs1((long)N * Kp), 256, 0, stream>>>(W, K, N, Kp, Wt);
    return Wt;
  };
  half_t* Tl1_0 = prep(Wl1_0,    6,   64,   32);
  half_t* Tl1_1 = prep(Wl1_1,   64,   64,   64);
  half_t* Tl1_2 = prep(Wl1_2,   64,  128,   64);
  half_t* Tg1   = prep(Wg1,    128,  128,  128);
  half_t* Tl2_0 = prep(Wl2_0,  131,  128,  160);
  half_t* Tl2_1 = prep(Wl2_1,  128,  128,  128);
  half_t* Tl2_2 = prep(Wl2_2,  128,  256,  128);
  half_t* Tg2   = prep(Wg2,    256,  256,  256);
  half_t* Tl3_0 = prep(Wl3_0,  259,  256,  288);
  half_t* Tl3_1 = prep(Wl3_1,  256,  512,  256);
  half_t* Tg3   = prep(Wg3,    512,  512,  512);
  half_t* Tgs_0 = prep(Wgs_0,  515,  512,  544);
  half_t* Tgs_1 = prep(Wgs_1,  512, 1024,  512);
  half_t* Tf_0  = prep(Wf_0,  1024,  512, 1024);
  half_t* Tf_1  = prep(Wf_1,   512,  256,  512);

  size_t bigE = (size_t)Bn*m1*KNB*128;           // 69.2M elems (138 MB f16)
  half_t* bufA = (half_t*)alloc(bigE * sizeof(half_t));
  half_t* bufB = (half_t*)alloc(bigE * sizeof(half_t));

  auto gemm = [&](const half_t* A, const half_t* Wt, const float* bb, half_t* Co,
                  int M, int N, int Kp, int relu) {
    dim3 g((unsigned)((M + 31) / 32), (unsigned)(N / 64));
    if ((M & 31) == 0)
      k_wmma_gemm<<<g, 128, 0, stream>>>(A, Wt, bb, Co, M, N, Kp, relu);
    else
      k_wmma_gemm_g<<<g, 128, 0, stream>>>(A, Wt, bb, Co, M, N, Kp, relu);
  };

  // ======== Stage 1: n=8192, m=2048, local [6,64,64,128], global [128,128] ==
  k_f32_to_f16<<<gs1((long)Bn*N0*3), 256, 0, stream>>>(x, xh0, (long)Bn*N0*3);
  k_fps<<<Bn, 256, 0, stream>>>(pos, N0, m1, idx1, dist);
  k_gather_cpos<<<gs1(Bn*m1), 256, 0, stream>>>(pos, idx1, N0, m1, Bn*m1, cpos1);
  k_knn<<<Bn*m1, 128, 0, stream>>>(pos, cpos1, N0, m1, nidx1, mask1);
  long rows1 = (long)Bn*m1*KNB;                                    // 540672
  k_gather_feat<<<gs1(rows1*32), 256, 0, stream>>>(xh0, 3, pos, cpos1, nidx1,
                                                   N0, m1, 32, bufA, rows1*32);
  gemm(bufA, Tl1_0, bl1_0, bufB, (int)rows1,  64,  32, 1);
  gemm(bufB, Tl1_1, bl1_1, bufA, (int)rows1,  64,  64, 1);
  gemm(bufA, Tl1_2, bl1_2, bufB, (int)rows1, 128,  64, 0);
  k_maxagg<<<gs1((long)Bn*m1*128), 256, 0, stream>>>(bufB, mask1, 128, (long)Bn*m1, bufA);
  gemm(bufA, Tg1, bg1, x2, Bn*m1, 128, 128, 0);

  // ======== Stage 2: n=2048, m=512, local [131,128,128,256], global [256,256]
  k_fps<<<Bn, 256, 0, stream>>>(cpos1, m1, m2, idx2, dist);
  k_gather_cpos<<<gs1(Bn*m2), 256, 0, stream>>>(cpos1, idx2, m1, m2, Bn*m2, cpos2);
  k_knn<<<Bn*m2, 128, 0, stream>>>(cpos1, cpos2, m1, m2, nidx2, mask2);
  long rows2 = (long)Bn*m2*KNB;                                    // 135168
  k_gather_feat<<<gs1(rows2*160), 256, 0, stream>>>(x2, 128, cpos1, cpos2, nidx2,
                                                    m1, m2, 160, bufA, rows2*160);
  gemm(bufA, Tl2_0, bl2_0, bufB, (int)rows2, 128, 160, 1);
  gemm(bufB, Tl2_1, bl2_1, bufA, (int)rows2, 128, 128, 1);
  gemm(bufA, Tl2_2, bl2_2, bufB, (int)rows2, 256, 128, 0);
  k_maxagg<<<gs1((long)Bn*m2*256), 256, 0, stream>>>(bufB, mask2, 256, (long)Bn*m2, bufA);
  gemm(bufA, Tg2, bg2, x3, Bn*m2, 256, 256, 0);

  // ======== Stage 3: n=512, m=128, local [259,256,512], global [512,512] ====
  k_fps<<<Bn, 256, 0, stream>>>(cpos2, m2, m3, idx3, dist);
  k_gather_cpos<<<gs1(Bn*m3), 256, 0, stream>>>(cpos2, idx3, m2, m3, Bn*m3, cpos3);
  k_knn<<<Bn*m3, 128, 0, stream>>>(cpos2, cpos3, m2, m3, nidx3, mask3);
  long rows3 = (long)Bn*m3*KNB;                                    // 33792
  k_gather_feat<<<gs1(rows3*288), 256, 0, stream>>>(x3, 256, cpos2, cpos3, nidx3,
                                                    m2, m3, 288, bufA, rows3*288);
  gemm(bufA, Tl3_0, bl3_0, bufB, (int)rows3, 256, 288, 1);
  gemm(bufB, Tl3_1, bl3_1, bufA, (int)rows3, 512, 256, 0);
  k_maxagg<<<gs1((long)Bn*m3*512), 256, 0, stream>>>(bufA, mask3, 512, (long)Bn*m3, bufB);
  gemm(bufB, Tg3, bg3, hg, Bn*m3, 512, 512, 0);

  // ======== GSA [515,512,1024], global max pool, final [1024,512,256] =======
  long rowsG = (long)Bn*m3;                                        // 1024
  k_concat_pos<<<gs1(rowsG*544), 256, 0, stream>>>(hg, cpos3, 512, 544, rowsG, bufA);
  gemm(bufA, Tgs_0, bgs_0, bufB, (int)rowsG,  512, 544, 1);
  gemm(bufB, Tgs_1, bgs_1, bufA, (int)rowsG, 1024, 512, 0);
  k_maxpool<<<gs1(Bn*1024), 256, 0, stream>>>(bufA, m3, 1024, Bn, pooled);
  gemm(pooled, Tf_0, bf_0, t1, Bn, 512, 1024, 1);   // M=8 -> guarded kernel
  gemm(t1,     Tf_1, bf_1, t2, Bn, 256,  512, 0);   // M=8 -> guarded kernel
  k_f16_to_f32<<<gs1(Bn*256), 256, 0, stream>>>(t2, (float*)d_out, (long)Bn*256);
}
